// Multi_Head_Attention_58076547776580
// MI455X (gfx1250) — compile-verified, matrix operand order checked by reading, not run on hardware
//
#include <hip/hip_runtime.h>
#include <hip/hip_bf16.h>

// ---------------------------------------------------------------------------
// Multi-head attention, CDNA5 (gfx1250), wave32, bf16 WMMA + async-to-LDS.
//   D=1024, H=16, hd=64, B=2, L=2048.  Softmax over the HEADS axis (faithful
//   to the reference).  Attention fuses all 16 heads in one workgroup (one
//   head per wave), streaming 32-key tiles with double-buffered
//   global_load_async_to_lds_b128 staging (ASYNCcnt pipeline).
//   LDS buffers are selected by *offset* (not pointer tables) so reads stay
//   in addrspace(3) and lower to ds_load_b128, not flat_load.
// ---------------------------------------------------------------------------

typedef __attribute__((ext_vector_type(16))) __bf16 v16bf;
typedef __attribute__((ext_vector_type(8)))  float  v8f;

#define D_MODEL 1024
#define N_HEADS 16
#define HEAD_DIM 64
#define L_SEQ   2048
#define B_SZ    2
#define KT      32            // key tile
#define KV_STRIDE 1032        // padded LDS row stride (halves): bank-spread
#define TILE_E  (KT * KV_STRIDE)   // halves per K or V tile buffer
#define EST     33            // padded energy row stride (floats)
#define N_TILES (L_SEQ / KT)  // 64

// Map fragment element index -> K index for 16-bit A/B fragments (wave32):
// VGPR0..3 hold K 0..7 (lanes 0-15) / 8..15 (lanes 16-31); VGPR4..7 hold +16.
__host__ __device__ constexpr int kmap(int e, int half) {
    int v = e >> 1, p = e & 1;
    int base = (v < 4) ? (v * 2 + p) : (16 + (v - 4) * 2 + p);
    return base + (half ? 8 : 0);
}

// --- CDNA5 async global->LDS (per-lane 16B, tracked with ASYNCcnt) ---------
__device__ __forceinline__ void async_b128(unsigned lds_off, const void* gptr) {
    asm volatile("global_load_async_to_lds_b128 %0, %1, off"
                 :: "v"(lds_off), "v"(gptr) : "memory");
}
__device__ __forceinline__ void wait_async(void) {          // ASYNCcnt <= 0
    asm volatile("s_wait_asynccnt 0x0" ::: "memory");
}
__device__ __forceinline__ void wait_async_16(void) {       // ASYNCcnt <= 16
    asm volatile("s_wait_asynccnt 0x10" ::: "memory");
}

// ---------------------------------------------------------------------------
// One-time fp32 -> bf16 conversion (weights) so GEMM B-tiles async-copy.
// ---------------------------------------------------------------------------
__global__ __launch_bounds__(256)
void cvt_f32_bf16(const float* __restrict__ s, __bf16* __restrict__ d, int n4) {
    int i = blockIdx.x * 256 + threadIdx.x;
    if (i < n4) {
        float4 f = ((const float4*)s)[i];
        d[i*4+0] = (__bf16)f.x; d[i*4+1] = (__bf16)f.y;
        d[i*4+2] = (__bf16)f.z; d[i*4+3] = (__bf16)f.w;
    }
}

// ---------------------------------------------------------------------------
// GEMM + bias: C[M,N] = A[M,K] @ W[K,N] + bias.  A fp32 or bf16 (VGPR-staged,
// dtype conversion), W bf16 (async LDS copy).  64x128 tile, 8 waves, BK=32.
// ---------------------------------------------------------------------------
template <bool IN_BF16, bool OUT_F32>
__global__ __launch_bounds__(256)
void gemm_bias_kernel(const void* __restrict__ Ain, const __bf16* __restrict__ Wb,
                      const float* __restrict__ bias, void* __restrict__ Cout,
                      int M, int N, int K) {
    constexpr int AST = 40;    // A LDS stride (halves)
    constexpr int WST = 136;   // W LDS stride (halves) = 17 uint4
    __shared__ __bf16 aLds[64 * AST];
    __shared__ __bf16 wLds[32 * WST];

    const int tid  = threadIdx.x;
    const int lane = tid & 31;
    const int wv   = tid >> 5;
    const int m    = lane & 15;
    const int half = lane >> 4;
    const int wm   = wv & 3;     // 4 waves along M (16 rows each)
    const int wn   = wv >> 2;    // 2 waves along N (64 cols each)
    const int row0 = blockIdx.y * 64;
    const int col0 = blockIdx.x * 128;
    const unsigned wbase = (unsigned)(size_t)&wLds[0];

    v8f acc[4] = {};

    for (int k0 = 0; k0 < K; k0 += 32) {
        // W tile 32x128 bf16: pure async copy, 512 x b128, 2 per thread
        {
            const uint4* wsrc = (const uint4*)(Wb + (size_t)k0 * N + col0);
#pragma unroll
            for (int t = 0; t < 2; ++t) {
                int i = tid + t * 256;
                int r = i >> 4, c = i & 15;
                async_b128(wbase + (unsigned)(r * 17 + c) * 16u,
                           wsrc + (size_t)r * (N >> 3) + c);
            }
        }
        // A tile 64x32: dtype conversion forces the VGPR path
        for (int i = tid; i < 64 * 32; i += 256) {
            int r = i >> 5, c = i & 31;
            float v;
            if (IN_BF16) v = (float)((const __bf16*)Ain)[(size_t)(row0 + r) * K + k0 + c];
            else         v = ((const float*)Ain)[(size_t)(row0 + r) * K + k0 + c];
            aLds[r * AST + c] = (__bf16)v;
        }
        wait_async();
        __syncthreads();

        v16bf af;  // A fragment 16x32: lane = row, element -> K
#pragma unroll
        for (int e = 0; e < 16; ++e)
            af[e] = aLds[(wm * 16 + m) * AST + kmap(e, half)];

#pragma unroll
        for (int j = 0; j < 4; ++j) {
            v16bf bf;  // B fragment 32x16: lane = col, element -> K
#pragma unroll
            for (int e = 0; e < 16; ++e)
                bf[e] = wLds[kmap(e, half) * WST + wn * 64 + j * 16 + m];
            acc[j] = __builtin_amdgcn_wmma_f32_16x16x32_bf16(
                false, af, false, bf, (short)0, acc[j], false, false);
        }
        __syncthreads();
    }

    // Epilogue: C layout M = r + 8*half, N = lane&15
#pragma unroll
    for (int j = 0; j < 4; ++j)
#pragma unroll
        for (int r = 0; r < 8; ++r) {
            int row = row0 + wm * 16 + r + half * 8;
            int col = col0 + wn * 64 + j * 16 + m;
            float v = acc[j][r] + bias[col];
            if (OUT_F32) ((float*)Cout)[(size_t)row * N + col] = v;
            else         ((__bf16*)Cout)[(size_t)row * N + col] = (__bf16)v;
        }
}

// ---------------------------------------------------------------------------
// Fused attention with head-axis softmax.
// Grid: (L/16, B). Block: 512 = 16 waves, wave h == head h.
// K/V 32-key tiles (32 x 1024 bf16, shared by all heads) double-buffered in
// LDS via async copies; next tile streams while current tile is in WMMAs.
// LDS layout (halves): [ K0 | K1 | V0 | V1 | e_lds | rmax | rden ]
// ---------------------------------------------------------------------------
__global__ __launch_bounds__(512)
void attn_kernel(const __bf16* __restrict__ Qb, const __bf16* __restrict__ Kb,
                 const __bf16* __restrict__ Vb, const int* __restrict__ mask,
                 __bf16* __restrict__ Xb) {
    extern __shared__ char smem[];
    __bf16* kv_lds = (__bf16*)smem;                     // 4 tile buffers
    float*  e_lds  = (float*)(kv_lds + 4 * TILE_E);     // 16 heads x 16 q x 33
    float*  rmax   = e_lds + N_HEADS * 16 * EST;        // 16 x 33
    float*  rden   = rmax + 16 * EST;                   // 16 x 33

    const int tid  = threadIdx.x;
    const int h    = tid >> 5;     // head == wave
    const int lane = tid & 31;
    const int m    = lane & 15;
    const int half = lane >> 4;
    const int b    = blockIdx.y;
    const int q0   = blockIdx.x * 16;
    const float scale = 0.125f;    // 1/sqrt(64)

    const unsigned lds_base = (unsigned)(size_t)kv_lds;
    const unsigned TILE_BYTES = TILE_E * 2u;            // 66048

    // Q A-fragments for this head: 16 q-rows x 64 dims -> 2 frags (K=32 each)
    v16bf qa[2];
    const __bf16* qrow = Qb + ((size_t)(b * L_SEQ + q0 + m)) * D_MODEL + h * HEAD_DIM;
#pragma unroll
    for (int kk = 0; kk < 2; ++kk)
#pragma unroll
        for (int e = 0; e < 16; ++e)
            qa[kk][e] = qrow[kk * 32 + kmap(e, half)];

    v8f oacc[4] = {};  // 16q x 64d fp32 accumulators

    // Async prefetch of one K/V tile: 4096+4096 x b128 over 512 threads
    // -> 16 async instructions per lane (ASYNCcnt += 16 per wave).
    auto prefetch_kv = [&](int kt, int buf) {
        const uint4* ks = (const uint4*)(Kb + ((size_t)(b * L_SEQ + kt * KT)) * D_MODEL);
        const uint4* vs = (const uint4*)(Vb + ((size_t)(b * L_SEQ + kt * KT)) * D_MODEL);
        const unsigned kd = lds_base + (unsigned)buf * TILE_BYTES;
        const unsigned vd = lds_base + (unsigned)(2 + buf) * TILE_BYTES;
#pragma unroll
        for (int t = 0; t < 8; ++t) {
            int i = tid + t * 512;
            int r = i >> 7, c = i & 127;
            unsigned off = (unsigned)(r * 129 + c) * 16u;  // padded rows
            async_b128(kd + off, ks + i);
            async_b128(vd + off, vs + i);
        }
    };

    prefetch_kv(0, 0);

    for (int kt = 0; kt < N_TILES; ++kt) {
        const int cur = kt & 1;
        if (kt + 1 < N_TILES) { prefetch_kv(kt + 1, cur ^ 1); wait_async_16(); }
        else                  { wait_async(); }
        __syncthreads();
        // Offset-select the live buffers (keeps addrspace(3) => ds_load)
        const int ko = cur * TILE_E;
        const int vo = (2 + cur) * TILE_E;

        // Energy E = Q K^T for this head: two 16-key subtiles, hd=64 -> 2 k-steps
        v8f eacc[2] = {};
#pragma unroll
        for (int s = 0; s < 2; ++s) {
#pragma unroll
            for (int kk = 0; kk < 2; ++kk) {
                v16bf bf;  // B = K^T: lane = key (n), element -> dim (K)
#pragma unroll
                for (int e = 0; e < 16; ++e)
                    bf[e] = kv_lds[ko + (s * 16 + m) * KV_STRIDE + h * HEAD_DIM + kk * 32 + kmap(e, half)];
                eacc[s] = __builtin_amdgcn_wmma_f32_16x16x32_bf16(
                    false, qa[kk], false, bf, (short)0, eacc[s], false, false);
            }
        }

        // scale, mask (where mask==0 -> -1e10), publish to LDS
#pragma unroll
        for (int s = 0; s < 2; ++s)
#pragma unroll
            for (int r = 0; r < 8; ++r) {
                int ql = r + half * 8;
                int kl = s * 16 + m;
                float v = eacc[s][r] * scale;
                if (mask[(size_t)b * L_SEQ * L_SEQ + (size_t)(q0 + ql) * L_SEQ + kt * KT + kl] == 0)
                    v = -1e10f;
                e_lds[(h * 16 + ql) * EST + kl] = v;
            }
        __syncthreads();

        // Softmax over HEADS per (q,k): 512 threads <-> 512 elements
        {
            int q = tid >> 5, k = tid & 31;
            float mx = -3.4e38f;
#pragma unroll
            for (int hh = 0; hh < N_HEADS; ++hh)
                mx = fmaxf(mx, e_lds[(hh * 16 + q) * EST + k]);
            float sum = 0.f;
#pragma unroll
            for (int hh = 0; hh < N_HEADS; ++hh)
                sum += __expf(e_lds[(hh * 16 + q) * EST + k] - mx);
            rmax[q * EST + k] = mx;
            rden[q * EST + k] = 1.0f / sum;
        }
        __syncthreads();

        // Attention A-fragment (16q x 32k, bf16), then attn @ V
        v16bf af;
#pragma unroll
        for (int e = 0; e < 16; ++e) {
            int k = kmap(e, half);
            float ev = e_lds[(h * 16 + m) * EST + k];
            af[e] = (__bf16)(__expf(ev - rmax[m * EST + k]) * rden[m * EST + k]);
        }
#pragma unroll
        for (int j = 0; j < 4; ++j) {
            v16bf vf;  // B = V: lane = dim (n), element -> key (K)
#pragma unroll
            for (int e = 0; e < 16; ++e)
                vf[e] = kv_lds[vo + kmap(e, half) * KV_STRIDE + h * HEAD_DIM + j * 16 + m];
            oacc[j] = __builtin_amdgcn_wmma_f32_16x16x32_bf16(
                false, af, false, vf, (short)0, oacc[j], false, false);
        }
        __syncthreads();  // LDS reuse fence before next tile's compute
    }

    // Store X in head-concat layout [B, L, H*hd] (bf16) for the Wo GEMM
#pragma unroll
    for (int j = 0; j < 4; ++j)
#pragma unroll
        for (int r = 0; r < 8; ++r) {
            int ql = r + half * 8;
            Xb[((size_t)(b * L_SEQ + q0 + ql)) * D_MODEL + h * HEAD_DIM + j * 16 + m] =
                (__bf16)oacc[j][r];
        }
}

// ---------------------------------------------------------------------------
extern "C" void kernel_launch(void* const* d_in, const int* in_sizes, int n_in,
                              void* d_out, int out_size, void* d_ws, size_t ws_size,
                              hipStream_t stream) {
    const float* query = (const float*)d_in[0];
    const float* key   = (const float*)d_in[1];
    const float* value = (const float*)d_in[2];
    const int*   mask  = (const int*)  d_in[3];
    const float* Wq = (const float*)d_in[4];  const float* bq = (const float*)d_in[5];
    const float* Wk = (const float*)d_in[6];  const float* bk = (const float*)d_in[7];
    const float* Wv = (const float*)d_in[8];  const float* bv = (const float*)d_in[9];
    const float* Wo = (const float*)d_in[10]; const float* bo = (const float*)d_in[11];
    float* out = (float*)d_out;

    const int Mrows = B_SZ * L_SEQ;                       // 4096
    const size_t actE = (size_t)Mrows * D_MODEL;          // 4.19M halves
    const size_t wE   = (size_t)D_MODEL * D_MODEL;        // 1.05M halves
    // Workspace: Q,K,V,X bf16 (33.5 MB) + 4 bf16 weight copies (8.4 MB)
    __bf16* Qb  = (__bf16*)d_ws;
    __bf16* Kb  = Qb + actE;
    __bf16* Vb  = Kb + actE;
    __bf16* Xb  = Vb + actE;
    __bf16* Wqb = Xb + actE;
    __bf16* Wkb = Wqb + wE;
    __bf16* Wvb = Wkb + wE;
    __bf16* Wob = Wvb + wE;

    // One-time weight conversion so GEMM B-tiles are async-copyable bf16
    const int n4 = (int)(wE / 4);
    cvt_f32_bf16<<<n4 / 256, 256, 0, stream>>>(Wq, Wqb, n4);
    cvt_f32_bf16<<<n4 / 256, 256, 0, stream>>>(Wk, Wkb, n4);
    cvt_f32_bf16<<<n4 / 256, 256, 0, stream>>>(Wv, Wvb, n4);
    cvt_f32_bf16<<<n4 / 256, 256, 0, stream>>>(Wo, Wob, n4);

    dim3 gg(D_MODEL / 128, Mrows / 64);  // 8 x 64 tiles
    gemm_bias_kernel<false, false><<<gg, 256, 0, stream>>>(query, Wqb, bq, Qb, Mrows, D_MODEL, D_MODEL);
    gemm_bias_kernel<false, false><<<gg, 256, 0, stream>>>(key,   Wkb, bk, Kb, Mrows, D_MODEL, D_MODEL);
    gemm_bias_kernel<false, false><<<gg, 256, 0, stream>>>(value, Wvb, bv, Vb, Mrows, D_MODEL, D_MODEL);

    size_t attn_lds = (size_t)(4 * TILE_E) * sizeof(__bf16) +
                      (size_t)(N_HEADS * 16 * EST + 2 * 16 * EST) * sizeof(float); // ~302 KB
    dim3 ga(L_SEQ / 16, B_SZ);
    attn_kernel<<<ga, 512, attn_lds, stream>>>(Qb, Kb, Vb, mask, Xb);

    gemm_bias_kernel<true, true><<<gg, 256, 0, stream>>>(Xb, Wob, bo, out, Mrows, D_MODEL, D_MODEL);
}
